// SpectralGatingNetwork_23416161698243
// MI455X (gfx1250) — matrix operand order repaired and checked by reading, not compiled, not run on hardware
//
#include <hip/hip_runtime.h>
#include <hip/hip_bf16.h>

// Spectral gating: out = Re(IFFT(w * FFT(x))) along D=512, per (b,h,l) row.
// Tensor-core Cooley-Tukey FFT (512 = 16 x 32) on v_wmma_f32_16x16x32_f16.
// One wave per (h,l); b innermost so the 4KB w row is read from HBM once.
// Round 2: all LDS fragment traffic vectorized (b128/b64), interleaved complex
// scratch, transposed staging for B-side consumers, vectorized w gating.

typedef __attribute__((ext_vector_type(16))) _Float16 v16h;
typedef __attribute__((ext_vector_type(8)))  float    v8f;

#define Bc 8
#define Hc 8
#define Lc 2048
#define Dc 512
#define WAVES 4
#define PI_F 3.14159265358979323846f

__device__ __forceinline__ v8f wmma16(v16h a, v16h b, v8f c) {
  return __builtin_amdgcn_wmma_f32_16x16x32_f16(false, a, false, b, (short)0, c, false, false);
}

// ---- A-side gather from row-major interleaved complex scratch ----
// A-matrix 16x32 (16-bit): lane m = lane&15; K runs are {8g..8g+7} (e<8) and
// {16+8g..23+8g} (e>=8) -> two contiguous runs of 8 complex = 8 x b128 loads.
__device__ __forceinline__ void ldsA_cplx(const float* sc, int lane,
                                          v16h& Ar, v16h& Ai, v16h& Ani) {
  int g = (lane >> 4) & 1, m = lane & 15;
  const float4* p = (const float4*)(sc + (m * 32 + 8 * g) * 2);        // e = 0..7
  const float4* q = (const float4*)(sc + (m * 32 + 16 + 8 * g) * 2);   // e = 8..15
  float4 r[8];
#pragma unroll
  for (int j = 0; j < 4; ++j) { r[j] = p[j]; r[j + 4] = q[j]; }
#pragma unroll
  for (int j = 0; j < 8; ++j) {
    Ar[2 * j]     = (_Float16)r[j].x;
    Ai[2 * j]     = (_Float16)r[j].y;
    Ani[2 * j]    = (_Float16)(-r[j].y);
    Ar[2 * j + 1] = (_Float16)r[j].z;
    Ai[2 * j + 1] = (_Float16)r[j].w;
    Ani[2 * j + 1] = (_Float16)(-r[j].w);
  }
}

// ---- Stage-1 B fragment from transposed x (lxT[n2*16 + n1]) ----
// B 32x16: lane n = (lane&15)+16t, element e -> K = e + 16g (K>=16 zero-padded).
__device__ __forceinline__ v16h ldsBx_tr(const float* xt, int lane, int t) {
  int g = (lane >> 4) & 1, n = (lane & 15) + 16 * t;
  const float4* p = (const float4*)(xt + n * 16);
  float msk = g ? 0.0f : 1.0f;
  v16h b;
#pragma unroll
  for (int j = 0; j < 4; ++j) {
    float4 v4 = p[j];
    b[4 * j]     = (_Float16)(v4.x * msk);
    b[4 * j + 1] = (_Float16)(v4.y * msk);
    b[4 * j + 2] = (_Float16)(v4.z * msk);
    b[4 * j + 3] = (_Float16)(v4.w * msk);
  }
  return b;
}

// ---- IFFT-2 B fragments from transposed interleaved scratch scT[(n*16+K)*2] ----
__device__ __forceinline__ void ldsB_tr_cplx(const float* sc, int lane, int t,
                                             v16h& br, v16h& bi) {
  int g = (lane >> 4) & 1, n = (lane & 15) + 16 * t;
  const float4* p = (const float4*)(sc + n * 32);   // 16 complex, K = 0..15
  float msk = g ? 0.0f : 1.0f;                      // K>=16 padded to zero
#pragma unroll
  for (int j = 0; j < 8; ++j) {
    float4 q = p[j];
    br[2 * j]     = (_Float16)(q.x * msk);
    bi[2 * j]     = (_Float16)(q.y * msk);
    br[2 * j + 1] = (_Float16)(q.z * msk);
    bi[2 * j + 1] = (_Float16)(q.w * msk);
  }
}

// ---- Forward twiddle (* (c - i s)) + row-major interleaved scatter (b64) ----
__device__ __forceinline__ void tw_store_rm(float* sc, const float* tw,
                                            v8f dr, v8f di, int lane, int t) {
  int g = lane >> 4, n = (lane & 15) + 16 * t;
#pragma unroll
  for (int v = 0; v < 8; ++v) {
    int idx = (v + 8 * g) * 32 + n;
    float2 cs = ((const float2*)tw)[idx];
    float yr = dr[v], yi = di[v];
    float2 o;
    o.x = yr * cs.x + yi * cs.y;
    o.y = yi * cs.x - yr * cs.y;
    ((float2*)sc)[idx] = o;
  }
}

// ---- Inverse twiddle (* (c + i s)) + TRANSPOSED interleaved store (4 x b128) ----
__device__ __forceinline__ void tw_store_tr(float* sc, const float* tw,
                                            v8f dr, v8f di, int lane, int t) {
  int g = lane >> 4, n = (lane & 15) + 16 * t;
  float4 o[4];
#pragma unroll
  for (int j = 0; j < 4; ++j) {
    int v0 = 2 * j, v1 = 2 * j + 1;
    float2 cs0 = ((const float2*)tw)[(v0 + 8 * g) * 32 + n];
    float2 cs1 = ((const float2*)tw)[(v1 + 8 * g) * 32 + n];
    o[j].x = dr[v0] * cs0.x - di[v0] * cs0.y;
    o[j].y = di[v0] * cs0.x + dr[v0] * cs0.y;
    o[j].z = dr[v1] * cs1.x - di[v1] * cs1.y;
    o[j].w = di[v1] * cs1.x + dr[v1] * cs1.y;
  }
  float4* dst = (float4*)(sc + n * 32 + 16 * g);    // complex idx n*16 + (8g..8g+7)
#pragma unroll
  for (int j = 0; j < 4; ++j) dst[j] = o[j];
}

// ---- Gate with w (vector float4 loads: bins k consecutive in v) + rm scatter ----
__device__ __forceinline__ void gate_store_rm(float* sc, const float* lw,
                                              v8f zr, v8f zi, int lane, int t) {
  int g = lane >> 4, n = lane & 15;
  int k0 = 8 * g + 16 * (n + 16 * t);               // bin for v = 0
  const float4* wp = (const float4*)(lw + 2 * k0);  // 8 complex weights
#pragma unroll
  for (int j = 0; j < 4; ++j) {
    float4 wv = wp[j];
    int v0 = 2 * j, v1 = 2 * j + 1;
    int idx0 = (v0 + 8 * g) * 32 + n + 16 * t;
    int idx1 = idx0 + 32;
    float2 o0, o1;
    o0.x = wv.x * zr[v0] - wv.y * zi[v0];
    o0.y = wv.x * zi[v0] + wv.y * zr[v0];
    o1.x = wv.z * zr[v1] - wv.w * zi[v1];
    o1.y = wv.z * zi[v1] + wv.w * zr[v1];
    ((float2*)sc)[idx0] = o0;
    ((float2*)sc)[idx1] = o1;
  }
}

__device__ __forceinline__ void out_store(float* o, v8f d, int lane, int t) {
  int g = lane >> 4, n = (lane & 15) + 16 * t;
#pragma unroll
  for (int v = 0; v < 8; ++v)
    o[32 * (v + 8 * g) + n] = d[v];                 // out[32*m1 + m2]
}

__global__ __launch_bounds__(WAVES * 32)
void sgn_fft_wmma_kernel(const float* __restrict__ x,
                         const float* __restrict__ w,
                         float* __restrict__ out) {
  const int tid  = threadIdx.x;
  const int wave = tid >> 5;
  const int lane = tid & 31;
  const int wid  = blockIdx.x * WAVES + wave;       // 0 .. H*L-1
  const int h    = wid >> 11;
  const int l    = wid & (Lc - 1);

  // Per-wave slices: [xT/out 512][w 1024][cplx scratch 1024][twiddle(c,s) 1024]
  __shared__ __align__(16) float lds[WAVES][3584];
  float* lx  = &lds[wave][0];
  float* lw  = &lds[wave][512];
  float* lsc = &lds[wave][1536];
  float* twi = &lds[wave][2560];

  // ---- DFT matrix fragments in registers (once per wave) ----
  v16h C16a, S16a, Cinv, Sninv;
  {
    int g = lane >> 4, m = lane & 15;
#pragma unroll
    for (int e = 0; e < 16; ++e) {
      int r = e >> 1, hh = e & 1;
      int K = e + 16 * g; (void)r; (void)hh;
      float c = 0.0f, s = 0.0f;
      if (K < 16) {
        float ang = (2.0f * PI_F / 16.0f) * (float)((m * K) & 15);
        float sv, cv; __sincosf(ang, &sv, &cv);
        c = cv; s = -sv;                            // e^{-i ang}
      }
      C16a[e]  = (_Float16)c;
      S16a[e]  = (_Float16)s;
      Cinv[e]  = (_Float16)(c * (1.0f / 512.0f));
      Sninv[e] = (_Float16)(s * (1.0f / 512.0f));
    }
  }
  v16h Br0, Bi0, Br1, Bi1, cr0, ci0, cr1, ci1;
  {
    int g = lane >> 4, nn = lane & 15;
#pragma unroll
    for (int e = 0; e < 16; ++e) {
      int K = (e < 8) ? (e + 8 * g) : (16 + 8 * g + (e - 8));
      float a0 = (2.0f * PI_F / 32.0f) * (float)((K * nn) & 31);
      float a1 = (2.0f * PI_F / 32.0f) * (float)((K * (nn + 16)) & 31);
      float s0, c0, s1, c1;
      __sincosf(a0, &s0, &c0);
      __sincosf(a1, &s1, &c1);
      Br0[e] = (_Float16)c0;  Bi0[e] = (_Float16)(-s0);
      Br1[e] = (_Float16)c1;  Bi1[e] = (_Float16)(-s1);
      cr0[e] = (_Float16)c0;  ci0[e] = (_Float16)(s0);
      cr1[e] = (_Float16)c1;  ci1[e] = (_Float16)(s1);
    }
  }

  // ---- Interleaved twiddle table (c,s) of W512^{k1*n2}, once per wave ----
  for (int idx = lane; idx < 512; idx += 32) {
    int k1 = idx >> 5, n2 = idx & 31;
    float ang = (2.0f * PI_F / 512.0f) * (float)((k1 * n2) & 511);
    float sv, cv; __sincosf(ang, &sv, &cv);
    ((float2*)twi)[idx] = make_float2(cv, sv);
  }

  // ---- Load (h,l) weight row once (coalesced float4) ----
  const float* wrow = w + ((size_t)(h * Lc + l)) * (size_t)(Dc * 2);
#pragma unroll
  for (int i = lane; i < 256; i += 32)
    ((float4*)lw)[i] = ((const float4*)wrow)[i];

  const v8f zacc = {};

  // ---- Batch loop: 8 x-rows reuse this (h,l) weight row ----
  for (int b = 0; b < Bc; ++b) {
    const size_t row = (((size_t)b * Hc + h) * Lc + l) * (size_t)Dc;
    const float* xrow = x + row;
    float* orow = out + row;
    if (b + 1 < Bc)
      __builtin_prefetch(x + row + (size_t)Hc * Lc * Dc, 0, 1);

    // Coalesced global b128 loads, transposed into LDS: lxT[n2*16 + n1].
#pragma unroll
    for (int i = lane; i < 128; i += 32) {
      float4 v4 = ((const float4*)xrow)[i];
      int n = 4 * i, col = n >> 5, rb = n & 31;
      lx[(rb + 0) * 16 + col] = v4.x;
      lx[(rb + 1) * 16 + col] = v4.y;
      lx[(rb + 2) * 16 + col] = v4.z;
      lx[(rb + 3) * 16 + col] = v4.w;
    }

    // Forward stage 1: Y[k1,n2] = DFT16 * x  (real input, b128 B gathers)
    v16h bx0 = ldsBx_tr(lx, lane, 0);
    v16h bx1 = ldsBx_tr(lx, lane, 1);
    v8f yr0 = wmma16(C16a, bx0, zacc);
    v8f yi0 = wmma16(S16a, bx0, zacc);
    v8f yr1 = wmma16(C16a, bx1, zacc);
    v8f yi1 = wmma16(S16a, bx1, zacc);
    tw_store_rm(lsc, twi, yr0, yi0, lane, 0);
    tw_store_rm(lsc, twi, yr1, yi1, lane, 1);

    // Forward stage 2: Z[k1,k2] = Y' * DFT32  (K = 32 exact)
    v16h Ar, Ai, Ani;
    ldsA_cplx(lsc, lane, Ar, Ai, Ani);
    v8f zr0 = wmma16(Ar, Br0, wmma16(Ani, Bi0, zacc));
    v8f zi0 = wmma16(Ar, Bi0, wmma16(Ai,  Br0, zacc));
    v8f zr1 = wmma16(Ar, Br1, wmma16(Ani, Bi1, zacc));
    v8f zi1 = wmma16(Ar, Bi1, wmma16(Ai,  Br1, zacc));

    // Gate with w (float4 weight loads), scatter G[k1][k2] interleaved.
    gate_store_rm(lsc, lw, zr0, zi0, lane, 0);
    gate_store_rm(lsc, lw, zr1, zi1, lane, 1);

    // Inverse stage 1: U[k1,m2] = G * conj(DFT32)
    ldsA_cplx(lsc, lane, Ar, Ai, Ani);
    v8f ur0 = wmma16(Ar, cr0, wmma16(Ani, ci0, zacc));
    v8f ui0 = wmma16(Ar, ci0, wmma16(Ai,  cr0, zacc));
    v8f ur1 = wmma16(Ar, cr1, wmma16(Ani, ci1, zacc));
    v8f ui1 = wmma16(Ar, ci1, wmma16(Ai,  cr1, zacc));
    tw_store_tr(lsc, twi, ur0, ui0, lane, 0);   // transposed for B-side consumer
    tw_store_tr(lsc, twi, ur1, ui1, lane, 1);

    // Inverse stage 2 (real part, 1/512 folded): out = Cinv*Ur' + Sninv*Ui'
    v16h bur0, bui0, bur1, bui1;
    ldsB_tr_cplx(lsc, lane, 0, bur0, bui0);
    ldsB_tr_cplx(lsc, lane, 1, bur1, bui1);
    v8f o0 = wmma16(Cinv, bur0, wmma16(Sninv, bui0, zacc));
    v8f o1 = wmma16(Cinv, bur1, wmma16(Sninv, bui1, zacc));

    // Stage row-major into LDS (x buffer is dead by now), b128 store out.
    out_store(lx, o0, lane, 0);
    out_store(lx, o1, lane, 1);
#pragma unroll
    for (int i = lane; i < 128; i += 32)
      ((float4*)orow)[i] = ((float4*)lx)[i];
  }
}

extern "C" void kernel_launch(void* const* d_in, const int* in_sizes, int n_in,
                              void* d_out, int out_size, void* d_ws, size_t ws_size,
                              hipStream_t stream) {
  const float* x = (const float*)d_in[0];        // (B,H,L,D) f32
  // d_in[1] = mask (int32), unused
  const float* w = (const float*)d_in[2];        // (H,L,D,2) f32
  float* out = (float*)d_out;                    // (B,H,L,D) f32
  (void)in_sizes; (void)n_in; (void)out_size; (void)d_ws; (void)ws_size;

  dim3 grid((Hc * Lc) / WAVES), block(WAVES * 32);
  sgn_fft_wmma_kernel<<<grid, block, 0, stream>>>(x, w, out);
}
